// CPADE_GAT_layer_66348654788675
// MI455X (gfx1250) — matrix operand order repaired and checked by reading, not compiled, not run on hardware
//
#include <hip/hip_runtime.h>
#include <hip/hip_bf16.h>

typedef __attribute__((ext_vector_type(16))) _Float16 v16h;
typedef __attribute__((ext_vector_type(4)))  _Float16 v4h;
typedef __attribute__((ext_vector_type(8)))  float    v8f;

#define HD   128
#define NH   8
#define EHD  16
#define NB   64

__device__ __forceinline__ int   enc_f(float f){ int i=__float_as_int(f); return (i>=0)? i : (i ^ 0x7FFFFFFF); }
__device__ __forceinline__ float dec_f(int i){ return __int_as_float((i>=0)? i : (i ^ 0x7FFFFFFF)); }
__device__ __forceinline__ float elu_f(float z){ return z>0.f ? z : (__expf(z)-1.f); }

// ---------------------------------------------------------------- fills
__global__ void fill_f32(float* p, float v, long n){
  long i = (long)blockIdx.x*blockDim.x + threadIdx.x; if (i<n) p[i]=v;
}
__global__ void fill_i32(int* p, int v, long n){
  long i = (long)blockIdx.x*blockDim.x + threadIdx.x; if (i<n) p[i]=v;
}

// ---------------------------------------------------------------- WMMA GEMM
// C[M x N] = A[M x K](f32) * W[K x N](f32, cast->f16) + bias, f32 out.
// 256 threads = 8 wave32 waves; block covers 128 rows; wave w owns rows w*16..15.
// LDS slabs are stored in *fragment order* so each lane reads its whole 16-half
// fragment with contiguous 32-byte DS loads (no scalar gather). Double-buffered
// LDS + register prefetch pipelines global loads under the WMMAs.
template<int NT>
__global__ __launch_bounds__(256)
void gemm_bias_wmma(const float* __restrict__ A, const float* __restrict__ W,
                    const float* __restrict__ bias, float* __restrict__ C,
                    int M, int K){
  constexpr int N  = NT*16;
  constexpr int BT = N/128;                 // B staging tasks per thread (1 or 2)
  __shared__ _Float16 AsF[2][128*32];       // fragment-ordered: [wave][lane][h]
  __shared__ _Float16 BsF[2][NT*512];       // fragment-ordered: [nt][lane][h]

  const int t    = threadIdx.x;
  const int wave = t >> 5;
  const int lane = t & 31;
  const int m16  = lane & 15;
  const int hi   = lane >> 4;
  const int blockRow = blockIdx.x * 128;

  v8f acc[NT];
#pragma unroll
  for (int nt=0; nt<NT; ++nt)
#pragma unroll
    for (int r=0; r<8; ++r) acc[nt][r] = 0.0f;

  float4 apre[4];
  v16h   bpre[BT];

  auto prefetch = [&](int k0){
    // A: 128 rows x 32 k, one float4 per task (1024 tasks, 4/thread), coalesced
#pragma unroll
    for (int j=0;j<4;++j){
      int id  = t + j*256;
      int row = id >> 3, kk = (id & 7)*4;
      int gr  = blockRow + row;
      if (gr < M) apre[j] = *(const float4*)(A + (long)gr*K + k0 + kk);
      else        apre[j] = make_float4(0.f,0.f,0.f,0.f);
    }
    // W: 32 x N; task = (column, 16-row k-group); coalesced across threads
#pragma unroll
    for (int j=0;j<BT;++j){
      int id  = t + j*256;
      int col = id % N, kkg = id / N;
      v16h hv;
#pragma unroll
      for (int h=0; h<16; ++h)
        hv[h] = (_Float16)W[(long)(k0 + kkg*8 + (h&7) + (h>>3)*16)*N + col];
      bpre[j] = hv;
    }
  };

  auto commit = [&](int p){
#pragma unroll
    for (int j=0;j<4;++j){
      int id  = t + j*256;
      int row = id >> 3, kk = (id & 7)*4;
      int w = row >> 4, m = row & 15;
      int h2 = (kk >> 3) & 1;                       // lane-half select
      int hb = (kk & 7) + ((kk & 16) ? 8 : 0);      // half base within fragment
      v4h hv;
      hv[0]=(_Float16)apre[j].x; hv[1]=(_Float16)apre[j].y;
      hv[2]=(_Float16)apre[j].z; hv[3]=(_Float16)apre[j].w;
      *(v4h*)&AsF[p][(w*32 + h2*16 + m)*16 + hb] = hv;
    }
#pragma unroll
    for (int j=0;j<BT;++j){
      int id  = t + j*256;
      int col = id % N, kkg = id / N;
      int nt = col >> 4, m = col & 15;
      *(v16h*)&BsF[p][(nt*32 + kkg*16 + m)*16] = bpre[j];
    }
  };

  const int nslabs = K >> 5;
  prefetch(0);
  int p = 0;
  for (int ks = 0; ks < nslabs; ++ks){
    commit(p);
    __syncthreads();
    if (ks+1 < nslabs) prefetch((ks+1)*32);   // global loads overlap the WMMAs

    v16h afrag = *(const v16h*)&AsF[p][(wave*32 + lane)*16];
#pragma unroll
    for (int nt = 0; nt < NT; ++nt){
      v16h bfrag = *(const v16h*)&BsF[p][(nt*32 + lane)*16];
      acc[nt] = __builtin_amdgcn_wmma_f32_16x16x32_f16(
          false, afrag, false, bfrag, (short)0, acc[nt], false, false);
    }
    p ^= 1;
  }
  // epilogue: C/D layout — VGPR r, lanes0-15: M=r, lanes16-31: M=8+r; N=lane&15
#pragma unroll
  for (int nt = 0; nt < NT; ++nt){
    int n  = nt*16 + m16;
    float bv = bias ? bias[n] : 0.0f;
#pragma unroll
    for (int r = 0; r < 8; ++r){
      int m = blockRow + wave*16 + r + 8*hi;
      if (m < M) C[(long)m*N + n] = acc[nt][r] + bv;
    }
  }
}

// ---------------------------------------------------------------- gated attention pooling
__global__ void gap_score(const float* __restrict__ feat, const float* __restrict__ Wg,
                          const float* __restrict__ bg, const int* __restrict__ gid,
                          long n, float* __restrict__ g, int* __restrict__ menc){
  long i = (long)blockIdx.x*blockDim.x + threadIdx.x; if (i>=n) return;
  const float* f = feat + i*HD;
  float s = bg[0];
#pragma unroll 4
  for (int d=0; d<HD; ++d) s += f[d]*Wg[d];
  g[i] = s;
  atomicMax(&menc[gid[i]], enc_f(s));
}
__global__ void gap_expsum(float* __restrict__ g, const int* __restrict__ gid,
                           const int* __restrict__ menc, float* __restrict__ gs, long n){
  long i = (long)blockIdx.x*blockDim.x + threadIdx.x; if (i>=n) return;
  int b = gid[i];
  float e = __expf(g[i] - dec_f(menc[b]));
  g[i] = e;
  atomicAdd(&gs[b], e);
}
__global__ void gap_wsum(const float* __restrict__ feat, const float* __restrict__ g,
                         const float* __restrict__ gs, const int* __restrict__ gid,
                         long n, float* __restrict__ outp){
  long i = (long)blockIdx.x*blockDim.x + threadIdx.x; if (i >= n*HD) return;
  long node = i >> 7; int d = (int)(i & 127);
  int b = gid[node];
  atomicAdd(&outp[b*HD + d], (g[node]/gs[b]) * feat[i]);
}

// ---------------------------------------------------------------- att_comb
__global__ void qcomb_k(const float* __restrict__ qc, const float* __restrict__ k1,
                        const float* __restrict__ k2, float* __restrict__ ac){
  int b = threadIdx.x; if (b >= NB) return;
  float s0=0.f, s1=0.f;
  for (int d=0; d<HD; ++d){ s0 += qc[d]*k1[b*HD+d]; s1 += qc[d]*k2[b*HD+d]; }
  float m = fmaxf(s0,s1);
  float e0=__expf(s0-m), e1=__expf(s1-m), t=e0+e1;
  ac[b*2]   = e0/t;
  ac[b*2+1] = e1/t;
}
__global__ void fh_k(const float* __restrict__ h0, const float* __restrict__ M2,
                     const float* __restrict__ ac, const int* __restrict__ gid,
                     long n, float* __restrict__ fh){
  long i = (long)blockIdx.x*blockDim.x + threadIdx.x; if (i >= n*HD) return;
  long node = i >> 7;
  int b = gid[node];
  fh[i] = h0[i]*ac[b*2] + M2[i]*ac[b*2+1];
}

// ---------------------------------------------------------------- per-node el/er
__global__ void el_er_k(const float* __restrict__ feat, const float* __restrict__ Qbuf,
                        const int* __restrict__ gid, long n,
                        float* __restrict__ el, float* __restrict__ er){
  long i = (long)blockIdx.x*blockDim.x + threadIdx.x; if (i >= n*NH) return;
  long node = i / NH; int h = (int)(i - node*NH);
  const float* f = feat + node*HD + h*EHD;
  const float* q = Qbuf + (long)gid[node]*(2*NH*EHD) + h*(2*EHD);
  float sl=0.f, sr=0.f;
#pragma unroll
  for (int d=0; d<EHD; ++d){ sl += f[d]*q[d]; sr += f[d]*q[EHD+d]; }
  el[i]=sl; er[i]=sr;
}

// ---------------------------------------------------------------- edge softmax + aggregation
__global__ void edge_max_k(const float* __restrict__ el, const float* __restrict__ er,
                           const int* __restrict__ src, const int* __restrict__ dst,
                           long E, float* __restrict__ ebuf, int* __restrict__ menc){
  long i = (long)blockIdx.x*blockDim.x + threadIdx.x; if (i >= E*NH) return;
  long e = i / NH; int h = (int)(i - e*NH);
  float v = el[(long)src[e]*NH + h] + er[(long)dst[e]*NH + h];
  v = (v > 0.f) ? v : 2.0f*v;          // leaky_relu, negative_slope = 2.0
  ebuf[i] = v;
  atomicMax(&menc[(long)dst[e]*NH + h], enc_f(v));
}
__global__ void edge_exp_k(float* __restrict__ ebuf, const int* __restrict__ dst,
                           const int* __restrict__ menc, float* __restrict__ ssum, long E){
  long i = (long)blockIdx.x*blockDim.x + threadIdx.x; if (i >= E*NH) return;
  long e = i / NH; int h = (int)(i - e*NH);
  float ex = __expf(ebuf[i] - dec_f(menc[(long)dst[e]*NH + h]));
  ebuf[i] = ex;
  atomicAdd(&ssum[(long)dst[e]*NH + h], ex);
}
__global__ void edge_norm_k(float* __restrict__ ebuf, const int* __restrict__ dst,
                            const float* __restrict__ ssum, long E){
  long i = (long)blockIdx.x*blockDim.x + threadIdx.x; if (i >= E*NH) return;
  long e = i / NH; int h = (int)(i - e*NH);
  ebuf[i] = ebuf[i] / ssum[(long)dst[e]*NH + h];
}
__global__ void agg_k(const float* __restrict__ a, const float* __restrict__ ft,
                      const int* __restrict__ src, const int* __restrict__ dst,
                      long E, float* __restrict__ agg){
  long i = (long)blockIdx.x*blockDim.x + threadIdx.x; if (i >= E*NH) return;
  long e = i / NH; int h = (int)(i - e*NH);
  float av = a[i];
  const float* fs = ft  + (long)src[e]*HD + h*EHD;
  float*       ag = agg + (long)dst[e]*HD + h*EHD;
  __builtin_prefetch(fs, 0, 1);
#pragma unroll
  for (int j=0; j<EHD; ++j) atomicAdd(ag + j, fs[j]*av);
}

// ---------------------------------------------------------------- batchnorm
__global__ void bn_partial(const float* __restrict__ Y, int M, int N, float* __restrict__ stats){
  int c = threadIdx.x;
  long r0 = (long)blockIdx.x * 1024;
  long r1 = r0 + 1024; if (r1 > M) r1 = M;
  float s=0.f, s2=0.f;
  for (long r=r0; r<r1; ++r){ float y = Y[r*N + c]; s += y; s2 += y*y; }
  atomicAdd(&stats[c], s); atomicAdd(&stats[N+c], s2);
}
__global__ void bn_finalize(const float* __restrict__ stats, int M, int N,
                            float* __restrict__ mu, float* __restrict__ rsig){
  int c = threadIdx.x;
  float m = stats[c] / (float)M;
  float v = stats[N+c] / (float)M - m*m;
  if (v < 0.f) v = 0.f;
  mu[c] = m; rsig[c] = rsqrtf(v + 1e-5f);
}
__global__ void bn_elu_k(float* __restrict__ Y, const float* __restrict__ mu,
                         const float* __restrict__ rsig, long M, int N){
  long i = (long)blockIdx.x*blockDim.x + threadIdx.x; if (i >= M*(long)N) return;
  int c = (int)(i % N);
  Y[i] = elu_f((Y[i]-mu[c])*rsig[c]);
}
__global__ void bn_elu_add_k(const float* __restrict__ Y, const float* __restrict__ mu,
                             const float* __restrict__ rsig, const float* __restrict__ hlast,
                             long M, int N, float* __restrict__ outp){
  long i = (long)blockIdx.x*blockDim.x + threadIdx.x; if (i >= M*(long)N) return;
  int c = (int)(i % N);
  outp[i] = hlast[i] + elu_f((Y[i]-mu[c])*rsig[c]);
}

// ---------------------------------------------------------------- PE/DE & loss
__global__ void pede_k(const float* __restrict__ a, const int* __restrict__ ids,
                       long cnt, float* __restrict__ acc){
  long i = (long)blockIdx.x*blockDim.x + threadIdx.x; if (i >= cnt*NH) return;
  long k = i / NH; int h = (int)(i - k*NH);
  atomicAdd(&acc[h], a[(long)ids[k]*NH + h]);
}
__global__ void loss_k(const float* __restrict__ PE, const float* __restrict__ DE,
                       const int* __restrict__ lenp, float* __restrict__ outp){
  float s = 0.f;
  for (int h=0; h<NH; ++h) s += (PE[h]-DE[h]);
  outp[0] = -(s / (float)lenp[0]) / (float)NH;
}

// ---------------------------------------------------------------- host orchestration
extern "C" void kernel_launch(void* const* d_in, const int* in_sizes, int n_in,
                              void* d_out, int out_size, void* d_ws, size_t ws_size,
                              hipStream_t stream){
  const float* h_da_last = (const float*)d_in[0];
  const float* h_q_last  = (const float*)d_in[1];
  const float* h_da_0    = (const float*)d_in[2];
  const float* h_q_0     = (const float*)d_in[3];
  const float* W_L       = (const float*)d_in[4];
  const float* b_L       = (const float*)d_in[5];
  const float* gateq_W   = (const float*)d_in[6];
  const float* gateq_b   = (const float*)d_in[7];
  const float* gate1_W   = (const float*)d_in[8];
  const float* gate1_b   = (const float*)d_in[9];
  const float* gate2_W   = (const float*)d_in[10];
  const float* gate2_b   = (const float*)d_in[11];
  const float* Q_comb    = (const float*)d_in[12];
  const float* Qf_W1     = (const float*)d_in[13];
  const float* Qf_b1     = (const float*)d_in[14];
  const float* Qf_W2     = (const float*)d_in[15];
  const float* Qf_b2     = (const float*)d_in[16];
  const float* end_W1    = (const float*)d_in[17];
  const float* end_b1    = (const float*)d_in[18];
  const float* end_W2    = (const float*)d_in[19];
  const float* end_b2    = (const float*)d_in[20];
  const int* src_da = (const int*)d_in[21];
  const int* dst_da = (const int*)d_in[22];
  const int* src_q  = (const int*)d_in[23];
  const int* dst_q  = (const int*)d_in[24];
  const int* gid_da = (const int*)d_in[25];
  const int* gid_q  = (const int*)d_in[26];
  const int* PEid   = (const int*)d_in[27];
  const int* DEid   = (const int*)d_in[28];
  const int* lenMn  = (const int*)d_in[29];

  const int nda = in_sizes[0]/HD, nq = in_sizes[1]/HD;
  const int eda = in_sizes[21],   eq = in_sizes[23];
  const int npe = in_sizes[27],   nde = in_sizes[28];
  const long nmax = (nda > nq) ? nda : nq;
  const long emax = (eda > eq) ? eda : eq;

  // ---- workspace layout (floats) ----
  float* WS = (float*)d_ws;
  size_t cur = 0;
  auto alloc = [&](size_t nf){ float* p = WS + cur; cur += nf; return p; };
  float* glq   = alloc((size_t)NB*HD);
  float* qact  = alloc((size_t)NB*256);
  float* Qbuf  = alloc((size_t)NB*256);
  float* k1    = alloc((size_t)NB*HD);
  float* k2    = alloc((size_t)NB*HD);
  float* acomb = alloc((size_t)NB*2);
  float* gscr  = alloc((size_t)nmax);
  int*   gm    = (int*)alloc(NB);
  float* gs    = alloc(NB);
  float* stats = alloc(512);
  float* mu    = alloc(256);
  float* rsig  = alloc(256);
  float* PE    = alloc(8);
  float* DE    = alloc(8);
  float* el    = alloc((size_t)nmax*NH);
  float* er    = alloc((size_t)nmax*NH);
  int*   em    = (int*)alloc((size_t)nmax*NH);
  float* es    = alloc((size_t)nmax*NH);
  float* abuf  = alloc((size_t)emax*NH);
  float* feat  = alloc((size_t)nmax*HD);   // reused as agg
  float* fh    = alloc((size_t)nmax*HD);
  float* ft    = alloc((size_t)nmax*HD);   // reused as y2
  float* y1    = alloc((size_t)nmax*256);
  (void)ws_size; (void)n_in; (void)out_size;

  float* out_da   = (float*)d_out;
  float* out_q    = out_da + (size_t)nda*HD;
  float* out_loss = out_q  + (size_t)nq*HD;

  auto cdiv = [](long a, long b){ return (unsigned)((a + b - 1)/b); };

  auto gemm = [&](const float* A, const float* Wt, const float* bb, float* Cc, int M, int K, int N){
    dim3 g(cdiv(M,128));
    if (N == 128) hipLaunchKernelGGL((gemm_bias_wmma<8>),  g, dim3(256), 0, stream, A, Wt, bb, Cc, M, K);
    else          hipLaunchKernelGGL((gemm_bias_wmma<16>), g, dim3(256), 0, stream, A, Wt, bb, Cc, M, K);
  };

  auto run_gap = [&](const float* Wg, const float* bg, const float* fptr,
                     const int* gid, long n, float* outp){
    fill_i32<<<1,64,0,stream>>>(gm, (int)0x80000000, NB);
    fill_f32<<<1,64,0,stream>>>(gs, 0.f, NB);
    fill_f32<<<cdiv(NB*HD,256),256,0,stream>>>(outp, 0.f, (long)NB*HD);
    gap_score <<<cdiv(n,256),256,0,stream>>>(fptr, Wg, bg, gid, n, gscr, gm);
    gap_expsum<<<cdiv(n,256),256,0,stream>>>(gscr, gid, gm, gs, n);
    gap_wsum  <<<cdiv(n*HD,256),256,0,stream>>>(fptr, gscr, gs, gid, n, outp);
  };

  auto run_bn = [&](float* Y, int M, int N){
    fill_f32<<<cdiv(2*N,256),256,0,stream>>>(stats, 0.f, 2L*N);
    bn_partial <<<cdiv(M,1024), N, 0, stream>>>(Y, M, N, stats);
    bn_finalize<<<1, N, 0, stream>>>(stats, M, N, mu, rsig);
  };

  auto run_branch = [&](const float* hlast, const float* tmp, const float* h0, const float* M2,
                        const int* src, const int* dst, const int* gid,
                        long n, long E, float* outp){
    // feat = tmp @ W_L + b_L  (WMMA)
    gemm(tmp, W_L, b_L, feat, (int)n, HD, HD);
    el_er_k<<<cdiv(n*NH,256),256,0,stream>>>(feat, Qbuf, gid, n, el, er);
    // fh = att_comb(h0, M2)
    run_gap(gate1_W, gate1_b, h0, gid, n, k1);
    run_gap(gate2_W, gate2_b, M2, gid, n, k2);
    qcomb_k<<<1,NB,0,stream>>>(Q_comb, k1, k2, acomb);
    fh_k<<<cdiv(n*HD,256),256,0,stream>>>(h0, M2, acomb, gid, n, fh);
    // ft = fh @ W_L + b_L  (WMMA)
    gemm(fh, W_L, b_L, ft, (int)n, HD, HD);
    // edge softmax over dst
    fill_i32<<<cdiv(n*NH,256),256,0,stream>>>(em, (int)0x80000000, n*NH);
    fill_f32<<<cdiv(n*NH,256),256,0,stream>>>(es, 0.f, n*NH);
    edge_max_k <<<cdiv(E*NH,256),256,0,stream>>>(el, er, src, dst, E, abuf, em);
    edge_exp_k <<<cdiv(E*NH,256),256,0,stream>>>(abuf, dst, em, es, E);
    edge_norm_k<<<cdiv(E*NH,256),256,0,stream>>>(abuf, dst, es, E);
    // agg = segment_sum(ft[src] * a, dst)   (agg overlays feat)
    fill_f32<<<cdiv(n*HD,256),256,0,stream>>>(feat, 0.f, n*HD);
    agg_k<<<cdiv(E*NH,256),256,0,stream>>>(abuf, ft, src, dst, E, feat);
    // end FNN: y1 = agg @ end_W1 ; elu(bn) in place ; y2 = y1 @ end_W2 (overlays ft)
    gemm(feat, end_W1, end_b1, y1, (int)n, HD, 256);
    run_bn(y1, (int)n, 256);
    bn_elu_k<<<cdiv(n*256,256),256,0,stream>>>(y1, mu, rsig, n, 256);
    gemm(y1, end_W2, end_b2, ft, (int)n, 256, HD);
    run_bn(ft, (int)n, HD);
    bn_elu_add_k<<<cdiv(n*HD,256),256,0,stream>>>(ft, mu, rsig, hlast, n, HD, outp);
  };

  // ---- Q pipeline ----
  run_gap(gateq_W, gateq_b, h_q_last, gid_q, nq, glq);
  gemm(glq, Qf_W1, Qf_b1, qact, NB, HD, 256);
  run_bn(qact, NB, 256);
  bn_elu_k<<<cdiv((long)NB*256,256),256,0,stream>>>(qact, mu, rsig, NB, 256);
  gemm(qact, Qf_W2, Qf_b2, Qbuf, NB, 256, 256);
  run_bn(Qbuf, NB, 256);
  bn_elu_k<<<cdiv((long)NB*256,256),256,0,stream>>>(Qbuf, mu, rsig, NB, 256);

  // ---- branches (q first so abuf holds a_da at the end) ----
  // q branch: tmp = h_q_last, M2 = h_q_last + (h_q_last - tmp) = h_q_last
  run_branch(h_q_last, h_q_last, h_q_0, h_q_last, src_q, dst_q, gid_q, nq, eq, out_q);
  run_branch(h_da_last, h_da_last, h_da_0, h_da_last, src_da, dst_da, gid_da, nda, eda, out_da);

  // ---- PE / DE / loss (uses a_da left in abuf) ----
  fill_f32<<<1,16,0,stream>>>(PE, 0.f, 16);   // PE and DE are contiguous
  pede_k<<<cdiv((long)npe*NH,256),256,0,stream>>>(abuf, PEid, npe, PE);
  pede_k<<<cdiv((long)nde*NH,256),256,0,stream>>>(abuf, DEid, nde, DE);
  loss_k<<<1,1,0,stream>>>(PE, DE, lenMn, out_loss);
}
